// KeystrokeClassificator_7258494730610
// MI455X (gfx1250) — compile-verified, matrix operand order checked by reading, not run on hardware
//
#include <hip/hip_runtime.h>
#include <hip/hip_bf16.h>

typedef __attribute__((ext_vector_type(16))) _Float16 v16h;
typedef __attribute__((ext_vector_type(8)))  _Float16 v8h;
typedef __attribute__((ext_vector_type(8)))  float    v8f;
typedef __attribute__((ext_vector_type(2)))  float    v2f;

#if __has_builtin(__builtin_amdgcn_wmma_f32_16x16x4_f32)
#define HAVE_WMMA_F32X4 1
#else
#define HAVE_WMMA_F32X4 0
#endif

namespace kc {
constexpr int Hd    = 32;     // d_model
constexpr int NHd   = 8;      // heads
constexpr int HDd   = 4;      // head dim
constexpr int FFd   = 2048;   // ffn dim
constexpr int NLd   = 6;      // layers
constexpr int LSEQ  = 4097;   // CLS + 4096
constexpr int LP    = 4112;   // padded to 257*16
constexpr int NT    = 257;    // 16-row tiles
constexpr int NPAIR = 129;    // 32-key pairs (covers 258 tiles, last masked)
constexpr int FRAG  = 512;    // halves per packed fragment (32 lanes x 16)
}
using namespace kc;

// K-dim index mapping for 16-bit A/B fragments (16x32 / 32x16), wave32:
// lanes 0-15 hold K {0..7, 16..23}; lanes 16-31 hold K {8..15, 24..31}.
__device__ __forceinline__ int kmap(int e, int half) {
    return e + 8 * half + (e & 8);
}

__device__ __forceinline__ v8f wmma16(v16h a, v16h b, v8f c) {
    return __builtin_amdgcn_wmma_f32_16x16x32_f16(
        false, a, false, b, (short)0, c, false, false);
}

// Packed fragment load: 32 contiguous bytes per lane (2x global_load_b128).
__device__ __forceinline__ v16h load_frag(const _Float16* __restrict__ P, int f, int lane) {
    return *(const v16h*)(P + ((size_t)f * 32 + lane) * 16);
}

// A fragment from row-major f32 (used only for the small attn-out matrix).
__device__ __forceinline__ v16h load_a16(const float* __restrict__ p0, int ld, int lane) {
    int half = lane >> 4, row = lane & 15;
    const float* p = p0 + row * ld;
    v16h a;
#pragma unroll
    for (int e = 0; e < 16; ++e) a[e] = (_Float16)p[kmap(e, half)];
    return a;
}

__device__ __forceinline__ float wave_sum(float v) {
#pragma unroll
    for (int off = 16; off > 0; off >>= 1) v += __shfl_xor(v, off, 32);
    return v;
}

// -------- one-time weight packing into f16 WMMA B-fragment layout ----------
__global__ void pack_w_kernel(const float* __restrict__ W, int ldw, int ntn,
                              _Float16* __restrict__ P) {
    int f = blockIdx.x, lane = threadIdx.x;
    int nt = f % ntn, kc = f / ntn;
    int half = lane >> 4, n = lane & 15;
    const float* p = W + (size_t)(nt * 16 + n) * ldw + kc * 32;
    _Float16* o = P + ((size_t)f * 32 + lane) * 16;
#pragma unroll
    for (int e = 0; e < 16; ++e) o[e] = (_Float16)p[kmap(e, half)];
}

// Pack x [LP x 32] f32 into f16 A-fragments (one fragment per 16-row tile).
__global__ void pack_x_kernel(const float* __restrict__ x, _Float16* __restrict__ xh) {
    int qt = blockIdx.x, lane = threadIdx.x;
    int half = lane >> 4, row = lane & 15;
    const float* p = x + (size_t)(qt * 16 + row) * Hd;
    _Float16* o = xh + ((size_t)qt * 32 + lane) * 16;
#pragma unroll
    for (int e = 0; e < 16; ++e) o[e] = (_Float16)p[kmap(e, half)];
}

// -------- per-layer K packing into the score-WMMA B operand layout ----------
// f32x4 path: 2 contiguous floats per lane (b64 load in the flash loop).
// fallback:   zero-padded f16 fragment (K=32, only head dims valid).
__global__ void pack_k_kernel(const float* __restrict__ qkv, void* __restrict__ Kp) {
    int kt = blockIdx.x, h = blockIdx.y, lane = threadIdx.x;
    int half = lane >> 4, n = lane & 15;
    const float* kp = qkv + (size_t)(kt * 16 + n) * 96 + 32 + h * HDd;
#if HAVE_WMMA_F32X4
    v2f v; v.x = kp[2 * half + 0]; v.y = kp[2 * half + 1];
    *((v2f*)Kp + ((size_t)h * NT + kt) * 32 + lane) = v;
#else
    _Float16* o = (_Float16*)Kp + (((size_t)h * NT + kt) * 32 + lane) * 16;
#pragma unroll
    for (int e = 0; e < 16; ++e) {
        int k = kmap(e, half);
        o[e] = (_Float16)((k < HDd) ? kp[k] : 0.0f);
    }
#endif
}

// -------- per-layer V packing into PV-WMMA B fragments (per head, 32-key pair) --
__global__ void pack_v_kernel(const float* __restrict__ qkv, _Float16* __restrict__ Vp) {
    int p = blockIdx.x, h = blockIdx.y, lane = threadIdx.x;
    int half = lane >> 4, n = lane & 15;
    _Float16* o = Vp + (((size_t)h * NPAIR + p) * 32 + lane) * 16;
#pragma unroll
    for (int e = 0; e < 16; ++e) {
        int k = kmap(e, half);
        int key = p * 32 + k;
        float vv = (n < HDd && key < LSEQ)
                     ? qkv[(size_t)key * 96 + 64 + h * HDd + n] : 0.0f;
        o[e] = (_Float16)vv;
    }
}

// ---------------- embedding + sinusoidal PE + CLS ----------------
__global__ void embed_kernel(const float* __restrict__ data,
                             const float* __restrict__ lin_w,
                             const float* __restrict__ lin_b,
                             float* __restrict__ x) {
    int row = blockIdx.x * blockDim.x + threadIdx.x;
    if (row >= LP) return;
    if (row == 0) {
        for (int c = 0; c < Hd; ++c) x[c] = -1.0f;
        return;
    }
    if (row >= LSEQ) {
        for (int c = 0; c < Hd; ++c) x[row * Hd + c] = 0.0f;
        return;
    }
    int s = row - 1;
    float ts = data[s * 3 + 0];
    float f0 = data[s * 3 + 1];
    float f1 = data[s * 3 + 2];
    float ti = (float)((int)(ts / 100.0f));
    for (int c = 0; c < Hd; ++c) {
        float v = fmaxf(f0 * lin_w[c * 2 + 0] + f1 * lin_w[c * 2 + 1] + lin_b[c], 0.0f);
        int j = c >> 1;
        float dv = __expf((float)(2 * j) * (-0.28782313662425574f)); // -ln(1e4)/32
        float ang = ti * dv;
        float pe = (c & 1) ? __cosf(ang) : __sinf(ang);
        x[row * Hd + c] = v + pe;
    }
}

// ------- QKV projection: one block per row tile, A reused across 6 WMMAs ------
__global__ void qkv_kernel(const _Float16* __restrict__ xh,
                           const _Float16* __restrict__ Wp,  // 6 packed frags
                           const float* __restrict__ bias,
                           float* __restrict__ qkv) {
    int qt = blockIdx.x, lane = threadIdx.x;
    int half = lane >> 4, n = lane & 15;
    v16h a = load_frag(xh, qt, lane);
#pragma unroll
    for (int nt = 0; nt < 6; ++nt) {
        v16h b = load_frag(Wp, nt, lane);
        v8f c = {};
        c = wmma16(a, b, c);
        int col = nt * 16 + n;
        float bv = bias[col];
#pragma unroll
        for (int r = 0; r < 8; ++r) {
            int rowg = qt * 16 + r + 8 * half;
            qkv[(size_t)rowg * 96 + col] = (rowg < LSEQ) ? (c[r] + bv) : 0.0f;
        }
    }
}

// ---------------- attention: flash 2-pass, register-resident softmax ----------
__global__ void attn_kernel(const float* __restrict__ qkv,
                            const void* __restrict__ Kp,
                            const _Float16* __restrict__ Vp,
                            float* __restrict__ att) {
    int qt = blockIdx.x, h = blockIdx.y, lane = threadIdx.x;
    int half = lane >> 4, n = lane & 15;

    __shared__ __align__(32) _Float16 pf[16][32];

    // Q fragment (pre-scaled by 1/sqrt(HD) = 0.5)
#if HAVE_WMMA_F32X4
    v2f aq;
    {
        const float* qp = qkv + (size_t)(qt * 16 + n) * 96 + h * HDd;
        aq.x = qp[2 * half + 0] * 0.5f;
        aq.y = qp[2 * half + 1] * 0.5f;
    }
#else
    v16h aq;
    {
        const float* qp = qkv + (size_t)(qt * 16 + n) * 96 + h * HDd;
#pragma unroll
        for (int e = 0; e < 16; ++e) {
            int k = kmap(e, half);
            aq[e] = (_Float16)((k < HDd) ? qp[k] * 0.5f : 0.0f);
        }
    }
#endif

    auto score = [&](int kt) -> v8f {
#if HAVE_WMMA_F32X4
        v2f bk = *((const v2f*)Kp + ((size_t)h * NT + kt) * 32 + lane);
        v8f z = {};
        return __builtin_amdgcn_wmma_f32_16x16x4_f32(
            false, aq, false, bk, (short)0, z, false, false);
#else
        v16h bk = load_frag((const _Float16*)Kp, h * NT + kt, lane);
        v8f z = {};
        return wmma16(aq, bk, z);
#endif
    };

    // ---- pass 1: online max/sum; (m,s) for 8 rows live in registers,
    //      replicated across the 16 lanes of each half-wave via shfl_xor. ----
    float rm[8], rs[8];
#pragma unroll
    for (int r = 0; r < 8; ++r) { rm[r] = -3.0e38f; rs[r] = 0.0f; }

    for (int kt = 0; kt < NT; ++kt) {
        if (kt + 1 < NT)
            __builtin_prefetch((const v2f*)Kp + ((size_t)h * NT + kt + 1) * 32 + lane, 0, 3);
        v8f d = score(kt);
        bool kvld = (kt * 16 + n) < LSEQ;
#pragma unroll
        for (int r = 0; r < 8; ++r) {
            float dm = kvld ? d[r] : -1.0e30f;
            float tm = dm;
#pragma unroll
            for (int mk = 1; mk < 16; mk <<= 1) tm = fmaxf(tm, __shfl_xor(tm, mk, 32));
            float mn = fmaxf(rm[r], tm);
            float e = __expf(dm - mn);
#pragma unroll
            for (int mk = 1; mk < 16; mk <<= 1) e += __shfl_xor(e, mk, 32);
            rs[r] = rs[r] * __expf(rm[r] - mn) + e;
            rm[r] = mn;
        }
    }

    // ---- pass 2: P = exp(score - m) (f16), O += P @ V, 32 keys per WMMA ----
    v8f acc = {};
    for (int kt = 0; kt < NT + 1; kt += 2) {      // 258 tiles; last is masked
        if (kt / 2 + 1 < NPAIR)
            __builtin_prefetch(Vp + (((size_t)h * NPAIR + kt / 2 + 1) * 32 + lane) * 16, 0, 3);
#pragma unroll
        for (int t = 0; t < 2; ++t) {
            int ktt = kt + t;
            if (ktt < NT) {
                v8f d = score(ktt);
                int key = ktt * 16 + n;
#pragma unroll
                for (int r = 0; r < 8; ++r) {
                    int m = r + 8 * half;
                    float p = (key < LSEQ) ? __expf(d[r] - rm[r]) : 0.0f;
                    pf[m][t * 16 + n] = (_Float16)p;
                }
            } else {
#pragma unroll
                for (int r = 0; r < 8; ++r) pf[r + 8 * half][t * 16 + n] = (_Float16)0.0f;
            }
        }
        __syncthreads();
        v16h ap;
        {
            int row = lane & 15;
            v8h lo = *(const v8h*)&pf[row][8 * half];
            v8h hi = *(const v8h*)&pf[row][16 + 8 * half];
#pragma unroll
            for (int e = 0; e < 8; ++e) { ap[e] = lo[e]; ap[e + 8] = hi[e]; }
        }
        v16h bv = load_frag(Vp, h * NPAIR + kt / 2, lane);
        acc = wmma16(ap, bv, acc);
        __syncthreads();
    }

    if (n < HDd) {
#pragma unroll
        for (int r = 0; r < 8; ++r) {
            int m = r + 8 * half;
            int qg = qt * 16 + m;
            float o = (qg < LSEQ) ? acc[r] / rs[r] : 0.0f;
            att[(size_t)qg * Hd + h * HDd + n] = o;
        }
    }
}

// ------- out-proj + residual + LN1; also emits packed f16 A-fragment of x -----
__global__ void proj_ln_kernel(const float* __restrict__ att,
                               const float* __restrict__ x,
                               const _Float16* __restrict__ Wp,  // 2 packed frags
                               const float* __restrict__ bias,
                               const float* __restrict__ g,
                               const float* __restrict__ be,
                               float* __restrict__ xout,
                               _Float16* __restrict__ xh) {
    int qt = blockIdx.x, lane = threadIdx.x;
    int half = lane >> 4, n = lane & 15;
    __shared__ float sh[16][33];

    v16h a = load_a16(att + (size_t)qt * 16 * Hd, Hd, lane);
#pragma unroll
    for (int nt = 0; nt < 2; ++nt) {
        v16h b = load_frag(Wp, nt, lane);
        v8f c = {};
        c = wmma16(a, b, c);
        int col = nt * 16 + n;
        float bv = bias[col];
#pragma unroll
        for (int r = 0; r < 8; ++r) {
            int m = r + 8 * half;
            sh[m][col] = c[r] + bv + x[(size_t)(qt * 16 + m) * Hd + col];
        }
    }
    __syncthreads();

    for (int m = 0; m < 16; ++m) {
        float v = sh[m][lane];
        float mu = wave_sum(v) * (1.0f / 32.0f);
        float dv = v - mu;
        float var = wave_sum(dv * dv) * (1.0f / 32.0f);
        float o = dv * rsqrtf(var + 1e-5f) * g[lane] + be[lane];
        int rowg = qt * 16 + m;
        o = (rowg < LSEQ) ? o : 0.0f;
        xout[(size_t)rowg * Hd + lane] = o;
        sh[m][lane] = o;
    }
    __syncthreads();
    {   // emit f16 A-fragment for the FFN stage
        int row = lane & 15;
        _Float16* o16 = xh + ((size_t)qt * 32 + lane) * 16;
#pragma unroll
        for (int e = 0; e < 16; ++e) o16[e] = (_Float16)sh[row][kmap(e, half)];
    }
}

// ------- FFN (FF1+ReLU+FF2 fused) + residual + LN2; emits next-layer frag -----
__global__ void ffn_kernel(const _Float16* __restrict__ xh,  // post-LN1 frags
                           const float* __restrict__ x,      // post-LN1 f32
                           const _Float16* __restrict__ W1p, // 128 packed frags
                           const float* __restrict__ b1,
                           const _Float16* __restrict__ W2p, // 128 packed frags
                           const float* __restrict__ b2,
                           const float* __restrict__ g,
                           const float* __restrict__ be,
                           float* __restrict__ xout,
                           _Float16* __restrict__ xhout) {
    int qt = blockIdx.x, lane = threadIdx.x;
    int half = lane >> 4, n = lane & 15;
    __shared__ __align__(32) _Float16 hh[16][32];
    __shared__ float sh[16][33];

    v16h ax = load_frag(xh, qt, lane);
    v8f acc0 = {}, acc1 = {};

    for (int c = 0; c < FFd / 32; ++c) {
#pragma unroll
        for (int t = 0; t < 2; ++t) {
            v16h b = load_frag(W1p, c * 2 + t, lane);
            v8f d = {};
            d = wmma16(ax, b, d);
            int col = t * 16 + n;
            float bb = b1[c * 32 + col];
#pragma unroll
            for (int r = 0; r < 8; ++r)
                hh[r + 8 * half][col] = (_Float16)fmaxf(d[r] + bb, 0.0f);
        }
        __syncthreads();
        v16h ah;
        {
            int row = lane & 15;
            v8h lo = *(const v8h*)&hh[row][8 * half];
            v8h hi = *(const v8h*)&hh[row][16 + 8 * half];
#pragma unroll
            for (int e = 0; e < 8; ++e) { ah[e] = lo[e]; ah[e + 8] = hi[e]; }
        }
        v16h bA = load_frag(W2p, c * 2 + 0, lane);
        v16h bB = load_frag(W2p, c * 2 + 1, lane);
        acc0 = wmma16(ah, bA, acc0);
        acc1 = wmma16(ah, bB, acc1);
        __syncthreads();
    }

#pragma unroll
    for (int r = 0; r < 8; ++r) {
        int m = r + 8 * half;
        const float* xr = x + (size_t)(qt * 16 + m) * Hd;
        sh[m][n]      = acc0[r] + b2[n]      + xr[n];
        sh[m][16 + n] = acc1[r] + b2[16 + n] + xr[16 + n];
    }
    __syncthreads();

    for (int m = 0; m < 16; ++m) {
        float v = sh[m][lane];
        float mu = wave_sum(v) * (1.0f / 32.0f);
        float dv = v - mu;
        float var = wave_sum(dv * dv) * (1.0f / 32.0f);
        float o = dv * rsqrtf(var + 1e-5f) * g[lane] + be[lane];
        int rowg = qt * 16 + m;
        o = (rowg < LSEQ) ? o : 0.0f;
        xout[(size_t)rowg * Hd + lane] = o;
        sh[m][lane] = o;
    }
    __syncthreads();
    {
        int row = lane & 15;
        _Float16* o16 = xhout + ((size_t)qt * 32 + lane) * 16;
#pragma unroll
        for (int e = 0; e < 16; ++e) o16[e] = (_Float16)sh[row][kmap(e, half)];
    }
}

// ---------------- classifier head ----------------
__global__ void final_kernel(const float* __restrict__ x,
                             const float* __restrict__ cls_w,
                             const float* __restrict__ cls_b,
                             float* __restrict__ out) {
    int lane = threadIdx.x;
    float v = x[lane] * cls_w[lane];
    v = wave_sum(v);
    if (lane == 0) out[0] = 1.0f / (1.0f + __expf(-(v + cls_b[0])));
}

extern "C" void kernel_launch(void* const* d_in, const int* in_sizes, int n_in,
                              void* d_out, int out_size, void* d_ws, size_t ws_size,
                              hipStream_t stream) {
    (void)in_sizes; (void)n_in; (void)out_size; (void)ws_size;
    const float* data  = (const float*)d_in[0];
    const float* lin_w = (const float*)d_in[1];
    const float* lin_b = (const float*)d_in[2];
    const float* qkv_w = (const float*)d_in[3];
    const float* qkv_b = (const float*)d_in[4];
    const float* out_w = (const float*)d_in[5];
    const float* out_b = (const float*)d_in[6];
    const float* ln1_g = (const float*)d_in[7];
    const float* ln1_b = (const float*)d_in[8];
    const float* ff1_w = (const float*)d_in[9];
    const float* ff1_b = (const float*)d_in[10];
    const float* ff2_w = (const float*)d_in[11];
    const float* ff2_b = (const float*)d_in[12];
    const float* ln2_g = (const float*)d_in[13];
    const float* ln2_b = (const float*)d_in[14];
    const float* cls_w = (const float*)d_in[15];
    const float* cls_b = (const float*)d_in[16];

    // f32 region (2,631,680 B, 32B-aligned end)
    float* x   = (float*)d_ws;                  // [LP x 32]
    float* qkv = x + (size_t)LP * Hd;           // [LP x 96]
    float* att = qkv + (size_t)LP * 96;         // [LP x 32]
    // f16 region (packed fragments, all offsets multiples of 32B)
    _Float16* xh    = (_Float16*)(att + (size_t)LP * Hd);   // NT frags
    _Float16* qkvWp = xh    + (size_t)NT * FRAG;            // 6 layers x 6
    _Float16* outWp = qkvWp + (size_t)NLd * 6 * FRAG;       // 6 layers x 2
    _Float16* ff1Wp = outWp + (size_t)NLd * 2 * FRAG;       // 6 layers x 128
    _Float16* ff2Wp = ff1Wp + (size_t)NLd * 128 * FRAG;     // 6 layers x 128
    void*     Kp    = (void*)(ff2Wp + (size_t)NLd * 128 * FRAG); // <= NHd*NT*FRAG halves
    _Float16* Vp    = (_Float16*)Kp + (size_t)NHd * NT * FRAG;   // NHd*NPAIR frags
    float* out = (float*)d_out;

    // one-time weight packing (f32 -> f16 fragment layout)
    for (int i = 0; i < NLd; ++i) {
        pack_w_kernel<<<6, 32, 0, stream>>>(
            qkv_w + (size_t)i * 3 * Hd * Hd, Hd, 6, qkvWp + (size_t)i * 6 * FRAG);
        pack_w_kernel<<<2, 32, 0, stream>>>(
            out_w + (size_t)i * Hd * Hd, Hd, 2, outWp + (size_t)i * 2 * FRAG);
        pack_w_kernel<<<128, 32, 0, stream>>>(
            ff1_w + (size_t)i * FFd * Hd, Hd, 128, ff1Wp + (size_t)i * 128 * FRAG);
        pack_w_kernel<<<128, 32, 0, stream>>>(
            ff2_w + (size_t)i * Hd * FFd, FFd, 2, ff2Wp + (size_t)i * 128 * FRAG);
    }

    embed_kernel<<<(LP + 63) / 64, 64, 0, stream>>>(data, lin_w, lin_b, x);
    pack_x_kernel<<<NT, 32, 0, stream>>>(x, xh);

    for (int i = 0; i < NLd; ++i) {
        qkv_kernel<<<NT, 32, 0, stream>>>(
            xh, qkvWp + (size_t)i * 6 * FRAG, qkv_b + (size_t)i * 3 * Hd, qkv);
        pack_k_kernel<<<dim3(NT, NHd), 32, 0, stream>>>(qkv, Kp);
        pack_v_kernel<<<dim3(NPAIR, NHd), 32, 0, stream>>>(qkv, Vp);
        attn_kernel<<<dim3(NT, NHd), 32, 0, stream>>>(qkv, Kp, Vp, att);
        proj_ln_kernel<<<NT, 32, 0, stream>>>(
            att, x, outWp + (size_t)i * 2 * FRAG, out_b + (size_t)i * Hd,
            ln1_g + (size_t)i * Hd, ln1_b + (size_t)i * Hd, x, xh);
        ffn_kernel<<<NT, 32, 0, stream>>>(
            xh, x, ff1Wp + (size_t)i * 128 * FRAG, ff1_b + (size_t)i * FFd,
            ff2Wp + (size_t)i * 128 * FRAG, ff2_b + (size_t)i * Hd,
            ln2_g + (size_t)i * Hd, ln2_b + (size_t)i * Hd, x, xh);
    }

    final_kernel<<<1, 32, 0, stream>>>(x, cls_w, cls_b, out);
}
